// TinyOnnGate_2379411882357
// MI455X (gfx1250) — matrix operand order repaired and checked by reading, not compiled, not run on hardware
//
#include <hip/hip_runtime.h>
#include <hip/hip_bf16.h>

typedef __attribute__((ext_vector_type(16))) __bf16 v16bf;
typedef __attribute__((ext_vector_type(8)))  float  v8f;

constexpr int N_TOKENS = 16384;
constexpr int HIDDEN   = 2048;
constexpr int NEXP     = 64;

// ---------------------------------------------------------------------------
// Kernel 1: normalize sim_matrix columns -> bf16 expert-major WnT[64][2048],
//           thr[e] = sigmoid(gates[e]).  One block per expert.
// ---------------------------------------------------------------------------
__global__ void prep_kernel(const float* __restrict__ W,      // [HIDDEN, NEXP]
                            const float* __restrict__ gates,  // [NEXP]
                            __bf16* __restrict__ WnT,         // [NEXP, HIDDEN]
                            float* __restrict__ thr) {        // [NEXP]
    const int e   = blockIdx.x;
    const int tid = threadIdx.x;          // 256 threads
    __shared__ float red[256];
    __shared__ float s_inv;

    float s = 0.f;
    for (int k = tid; k < HIDDEN; k += 256) {
        float v = W[(size_t)k * NEXP + e];
        s = fmaf(v, v, s);
    }
    red[tid] = s;
    __syncthreads();
    for (int o = 128; o > 0; o >>= 1) {
        if (tid < o) red[tid] += red[tid + o];
        __syncthreads();
    }
    if (tid == 0) {
        s_inv = 1.0f / fmaxf(sqrtf(red[0]), 1e-12f);
        thr[e] = 1.0f / (1.0f + expf(-gates[e]));
    }
    __syncthreads();
    const float inv = s_inv;
    for (int k = tid; k < HIDDEN; k += 256)
        WnT[(size_t)e * HIDDEN + k] = (__bf16)(W[(size_t)k * NEXP + e] * inv);
}

// ---------------------------------------------------------------------------
// Kernel 2: dots[t][e] = x[t,:] . Wn[:,e]  via v_wmma_f32_16x16x32_bf16.
//           One wave per 16-token M-tile, 4 N-tiles (64 experts).
//           Also computes invnorm[t] = 1/max(||x_t||, eps) inline.
// Block = 128 threads (4 waves) -> 64 rows per block, grid = 256 blocks.
// ---------------------------------------------------------------------------
__global__ void gemm_kernel(const float* __restrict__ x,      // [N_TOKENS, HIDDEN]
                            const __bf16* __restrict__ WnT,   // [NEXP, HIDDEN]
                            float* __restrict__ dots,         // [N_TOKENS, NEXP]
                            float* __restrict__ invnorm) {    // [N_TOKENS]
    const int lane    = threadIdx.x & 31;
    const int wave    = threadIdx.x >> 5;
    const int rowbase = blockIdx.x * 64 + wave * 16;
    const int r       = lane & 15;        // A: my row within tile; B: my column
    const int hi      = lane >> 4;        // 0 or 1
    const int aoff    = hi * 16;          // A K-offset (contiguous, matches B)
    const int boff    = hi * 16;          // B K-offset (contiguous, matches A)

    const float* xrow = x + (size_t)(rowbase + r) * HIDDEN;
    const __bf16* b0p = WnT + (size_t)(0 * 16 + r) * HIDDEN + boff;
    const __bf16* b1p = WnT + (size_t)(1 * 16 + r) * HIDDEN + boff;
    const __bf16* b2p = WnT + (size_t)(2 * 16 + r) * HIDDEN + boff;
    const __bf16* b3p = WnT + (size_t)(3 * 16 + r) * HIDDEN + boff;

    v8f acc0 = {}, acc1 = {}, acc2 = {}, acc3 = {};
    float ss = 0.f;

    for (int kk = 0; kk < HIDDEN; kk += 32) {
        // A fragment: lane<16 holds K {kk..kk+15},
        //             lane>=16 holds K {kk+16..kk+31}
        const float4* p = (const float4*)(xrow + kk + aoff);
        float4 a0 = p[0], a1 = p[1];
        const float4* q = (const float4*)(xrow + kk + 8 + aoff);
        float4 c0 = q[0], c1 = q[1];

        ss = fmaf(a0.x, a0.x, ss); ss = fmaf(a0.y, a0.y, ss);
        ss = fmaf(a0.z, a0.z, ss); ss = fmaf(a0.w, a0.w, ss);
        ss = fmaf(a1.x, a1.x, ss); ss = fmaf(a1.y, a1.y, ss);
        ss = fmaf(a1.z, a1.z, ss); ss = fmaf(a1.w, a1.w, ss);
        ss = fmaf(c0.x, c0.x, ss); ss = fmaf(c0.y, c0.y, ss);
        ss = fmaf(c0.z, c0.z, ss); ss = fmaf(c0.w, c0.w, ss);
        ss = fmaf(c1.x, c1.x, ss); ss = fmaf(c1.y, c1.y, ss);
        ss = fmaf(c1.z, c1.z, ss); ss = fmaf(c1.w, c1.w, ss);

        v16bf A;
        A[0]  = (__bf16)a0.x; A[1]  = (__bf16)a0.y;
        A[2]  = (__bf16)a0.z; A[3]  = (__bf16)a0.w;
        A[4]  = (__bf16)a1.x; A[5]  = (__bf16)a1.y;
        A[6]  = (__bf16)a1.z; A[7]  = (__bf16)a1.w;
        A[8]  = (__bf16)c0.x; A[9]  = (__bf16)c0.y;
        A[10] = (__bf16)c0.z; A[11] = (__bf16)c0.w;
        A[12] = (__bf16)c1.x; A[13] = (__bf16)c1.y;
        A[14] = (__bf16)c1.z; A[15] = (__bf16)c1.w;

        // B fragments: 16 contiguous bf16 of my expert column's K slice
        v16bf B0 = *(const v16bf*)(b0p + kk);
        v16bf B1 = *(const v16bf*)(b1p + kk);
        v16bf B2 = *(const v16bf*)(b2p + kk);
        v16bf B3 = *(const v16bf*)(b3p + kk);

        acc0 = __builtin_amdgcn_wmma_f32_16x16x32_bf16(false, A, false, B0,
                                                       (short)0, acc0, false, false);
        acc1 = __builtin_amdgcn_wmma_f32_16x16x32_bf16(false, A, false, B1,
                                                       (short)0, acc1, false, false);
        acc2 = __builtin_amdgcn_wmma_f32_16x16x32_bf16(false, A, false, B2,
                                                       (short)0, acc2, false, false);
        acc3 = __builtin_amdgcn_wmma_f32_16x16x32_bf16(false, A, false, B3,
                                                       (short)0, acc3, false, false);
    }

    // full row sumsq = my half + partner lane's half (lane ^ 16)
    ss += __shfl_xor(ss, 16);
    const float inv = 1.0f / fmaxf(sqrtf(ss), 1e-12f);
    if (lane < 16) invnorm[rowbase + lane] = inv;

    // C layout: lane holds N = r (+16*tile), rows hi*8+g in vgpr g
    float* drow = dots + (size_t)(rowbase + hi * 8) * NEXP + r;
    #pragma unroll
    for (int g = 0; g < 8; ++g) {
        float* dr = drow + (size_t)g * NEXP;
        dr[0]  = acc0[g];
        dr[16] = acc1[g];
        dr[32] = acc2[g];
        dr[48] = acc3[g];
    }
}

// ---------------------------------------------------------------------------
// Kernel 3: per-token epilogue. One wave per token, 2 experts per lane.
// Reads raw dots from out segment 2 (scratch), writes probs/pre/mask.
// ---------------------------------------------------------------------------
__global__ void epilogue_kernel(const float* __restrict__ thr,
                                const float* __restrict__ invnorm,
                                float* __restrict__ out) {
    const int lane = threadIdx.x & 31;
    const int t    = blockIdx.x * 8 + (threadIdx.x >> 5);

    float* probs = out;
    float* pre   = out + (size_t)N_TOKENS * NEXP;
    float* mask  = out + 2 * (size_t)N_TOKENS * NEXP;   // also holds raw dots

    const float2 d  = ((const float2*)(mask + (size_t)t * NEXP))[lane];
    const float2 th = ((const float2*)thr)[lane];
    const float inv = invnorm[t];

    const float L0 = d.x * inv, L1 = d.y * inv;       // cosine logits
    const float p0 = L0 - th.x, p1 = L1 - th.y;       // pre-activation
    const float g0 = fmaxf(p0, 0.f), g1 = fmaxf(p1, 0.f);
    float h0 = (p0 > 0.f) ? 1.f : 0.f;
    float h1 = (p1 > 0.f) ? 1.f : 0.f;

    float na = h0 + h1;
    #pragma unroll
    for (int m = 16; m > 0; m >>= 1) na += __shfl_xor(na, m);
    const bool inactive = (na == 0.f);

    float m0 = h0, m1 = h1;
    if (inactive) {
        // exact top-32 of logits via rank counting (index tie-break = topk order)
        int r0 = 0, r1 = 0;
        #pragma unroll
        for (int j = 0; j < 64; ++j) {
            float Lj = __shfl((j & 1) ? L1 : L0, j >> 1);
            r0 += (Lj > L0) || ((Lj == L0) && (j < 2 * lane));
            r1 += (Lj > L1) || ((Lj == L1) && (j < 2 * lane + 1));
        }
        m0 = (r0 < 32) ? 1.f : 0.f;
        m1 = (r1 < 32) ? 1.f : 0.f;
    }

    // masked softmax, fill = -FLT_MAX
    const float NEG = -__FLT_MAX__;
    float q0 = (m0 > 0.f) ? g0 : NEG;
    float q1 = (m1 > 0.f) ? g1 : NEG;
    float mx = fmaxf(q0, q1);
    #pragma unroll
    for (int m = 16; m > 0; m >>= 1) mx = fmaxf(mx, __shfl_xor(mx, m));
    float e0 = __expf(q0 - mx);
    float e1 = __expf(q1 - mx);
    float s = e0 + e1;
    #pragma unroll
    for (int m = 16; m > 0; m >>= 1) s += __shfl_xor(s, m);
    const float rs = 1.0f / s;

    ((float2*)(probs + (size_t)t * NEXP))[lane] = make_float2(e0 * rs, e1 * rs);
    ((float2*)(pre   + (size_t)t * NEXP))[lane] = make_float2(p0, p1);
    ((float2*)(mask  + (size_t)t * NEXP))[lane] = make_float2(m0, m1);
}

// ---------------------------------------------------------------------------
extern "C" void kernel_launch(void* const* d_in, const int* in_sizes, int n_in,
                              void* d_out, int out_size, void* d_ws, size_t ws_size,
                              hipStream_t stream) {
    const float* x     = (const float*)d_in[0];   // [16384, 2048]
    const float* simW  = (const float*)d_in[1];   // [2048, 64]
    const float* gates = (const float*)d_in[2];   // [64]
    float* out = (float*)d_out;                   // 3 x [16384, 64]

    // workspace layout (≈320 KB): WnT bf16 | invnorm f32 | thr f32
    __bf16* WnT     = (__bf16*)d_ws;
    float*  invnorm = (float*)((char*)d_ws + (size_t)NEXP * HIDDEN * sizeof(__bf16));
    float*  thr     = invnorm + N_TOKENS;

    // raw dot products staged in out segment 2 (overwritten by mask in epilogue)
    float* dots = out + 2 * (size_t)N_TOKENS * NEXP;

    prep_kernel<<<NEXP, 256, 0, stream>>>(simW, gates, WnT, thr);
    gemm_kernel<<<N_TOKENS / 64, 128, 0, stream>>>(x, WnT, dots, invnorm);
    epilogue_kernel<<<N_TOKENS / 8, 256, 0, stream>>>(thr, invnorm, out);
}
